// BiQRNN_24790551232670
// MI455X (gfx1250) — matrix operand order repaired and checked by reading, not compiled, not run on hardware
//
#include <hip/hip_runtime.h>

// ---------------- problem dims (fixed by the reference) ----------------
#define S_LEN 2048
#define BATCH 16
#define DIN0  512
#define HID   512
#define NCOL  1536              // 3*H
#define MROWS (S_LEN * BATCH)   // 32768
#define Y1C   1024              // 2*H
#define BK    32                // K-step (one bf16 WMMA)
#define LDP   40                // padded LDS row length (16B-aligned rows)

typedef __attribute__((ext_vector_type(16))) __bf16 v16bf;
typedef __attribute__((ext_vector_type(8)))  float  v8f;

// CDNA5 async global->LDS copy (ASYNCcnt-tracked, no VGPR round trip).
// GVS mode: mem_addr = SGPR base + 32-bit per-lane byte offset.
__device__ __forceinline__ void async_b128(unsigned lds_addr,
                                           unsigned long long sbase,
                                           unsigned voff) {
    asm volatile("global_load_async_to_lds_b128 %0, %1, %2"
                 :: "v"(lds_addr), "v"(voff), "s"(sbase) : "memory");
}
__device__ __forceinline__ void wait_async0() {
    asm volatile("s_wait_asynccnt 0" ::: "memory");
}

// ---------------------------------------------------------------------
// Convert + transpose X: [B,S,D] f32 -> time-major [S*B, D] bf16
// ---------------------------------------------------------------------
__global__ void __launch_bounds__(256)
cvt_x_kernel(const float* __restrict__ X, __bf16* __restrict__ Xb) {
    int idx = blockIdx.x * 256 + threadIdx.x;          // over MROWS*DIN0/4
    int d4  = idx % (DIN0 / 4);
    int m   = idx / (DIN0 / 4);
    int s = m / BATCH, b = m % BATCH;
    const float4 v = ((const float4*)(X + ((long)b * S_LEN + s) * DIN0))[d4];
    __bf16* o = Xb + (long)m * DIN0 + d4 * 4;
    o[0] = (__bf16)v.x; o[1] = (__bf16)v.y; o[2] = (__bf16)v.z; o[3] = (__bf16)v.w;
}

// ---------------------------------------------------------------------
// Convert + TRANSPOSE weights: W [K, NCOL] f32 -> Wt [NCOL, K] bf16.
// K-contiguous storage lets the GEMM stage B tiles with async b128 copies.
// ---------------------------------------------------------------------
__global__ void __launch_bounds__(256)
cvt_wt_kernel(const float* __restrict__ W, __bf16* __restrict__ Wt, int K) {
    __shared__ float tile[32][33];
    int bk = blockIdx.x;                 // K/32
    int bn = blockIdx.y;                 // NCOL/32
    int tx = threadIdx.x & 31, ty = threadIdx.x >> 5;   // 32 x 8
    #pragma unroll
    for (int r = 0; r < 32; r += 8)
        tile[ty + r][tx] = W[(long)(bk * 32 + ty + r) * NCOL + bn * 32 + tx];
    __syncthreads();
    #pragma unroll
    for (int r = 0; r < 32; r += 8)
        Wt[(long)(bn * 32 + ty + r) * K + bk * 32 + tx] = (__bf16)tile[tx][ty + r];
}

// ---------------------------------------------------------------------
// GEMM + bias + fused activation:
//   out[m,n] = act( sum_k A[m,k]*Wt[n,k] + bias[n] ),  act = tanh | sigmoid
// A: [M,K] bf16, Wt: [NCOL,K] bf16, out: [M,NCOL] f32.
// 128x128 tile / 8 waves / 4x2 16x16 accum per wave; double-buffered
// async global->LDS staging overlapped with WMMA.
// ---------------------------------------------------------------------
__global__ void __launch_bounds__(256)
gemm_act_kernel(const __bf16* __restrict__ A, const __bf16* __restrict__ Wt,
                const float* __restrict__ bias, float* __restrict__ out,
                int K) {
    __shared__ __bf16 As[2][128][LDP];
    __shared__ __bf16 Bs[2][128][LDP];

    const int tileM = blockIdx.x * 128;
    const int tileN = blockIdx.y * 128;
    const int tid   = threadIdx.x;
    const int wave  = tid >> 5;
    const int lane  = tid & 31;
    const int half  = lane >> 4;
    const int lrow  = lane & 15;
    const int wm    = wave & 1;      // 2 wave-rows (64 rows)
    const int wn    = wave >> 1;     // 4 wave-cols (32 cols)

    const unsigned asb = (unsigned)(unsigned long long)(void*)&As[0][0][0];
    const unsigned bsb = (unsigned)(unsigned long long)(void*)&Bs[0][0][0];
    const unsigned long long abase = (unsigned long long)A;
    const unsigned long long bbase = (unsigned long long)Wt;

    v8f zero = {};
    v8f acc[4][2];
    #pragma unroll
    for (int i = 0; i < 4; ++i)
        #pragma unroll
        for (int j = 0; j < 2; ++j) acc[i][j] = zero;

    // stage one 128xBK tile of A and of Wt into LDS buffer `buf`
    auto stage = [&](int buf, int kk) {
        #pragma unroll
        for (int i = 0; i < 2; ++i) {
            int c   = tid + i * 256;       // 512 b128 chunks per tile
            int row = c >> 2;              // 0..127
            int col = (c & 3) * 8;         // 0,8,16,24 (bf16 elems)
            unsigned lofs = (unsigned)((buf * 128 * LDP + row * LDP + col) * 2);
            unsigned ga = (unsigned)(((tileM + row) * K + kk + col) * 2);
            async_b128(asb + lofs, abase, ga);
            unsigned gb = (unsigned)(((tileN + row) * K + kk + col) * 2);
            async_b128(bsb + lofs, bbase, gb);
        }
    };

    stage(0, 0);
    wait_async0();
    __syncthreads();

    const int nk = K / BK;
    for (int t = 0; t < nk; ++t) {
        const int cur = t & 1;
        if (t + 1 < nk) stage(cur ^ 1, (t + 1) * BK);   // overlap next tile

        // ---- fragments per ISA 7.12.2 bf16 layouts (contiguous K pairs) ----
        v16bf afr[4], bfr[2];
        #pragma unroll
        for (int im = 0; im < 4; ++im) {
            int mrow = wm * 64 + im * 16 + lrow;
            #pragma unroll
            for (int v = 0; v < 8; ++v) {
                int k0 = (v < 4) ? (2 * v + 8 * half) : (2 * v + 8 + 8 * half);
                afr[im][2 * v]     = As[cur][mrow][k0];
                afr[im][2 * v + 1] = As[cur][mrow][k0 + 1];
            }
        }
        #pragma unroll
        for (int in = 0; in < 2; ++in) {
            int ncol = wn * 32 + in * 16 + lrow;
            #pragma unroll
            for (int v = 0; v < 8; ++v) {
                int k0 = (v < 4) ? (2 * v + 8 * half) : (2 * v + 8 + 8 * half);
                bfr[in][2 * v]     = Bs[cur][ncol][k0];
                bfr[in][2 * v + 1] = Bs[cur][ncol][k0 + 1];
            }
        }
        // ---- 8 WMMAs per K step ----
        #pragma unroll
        for (int im = 0; im < 4; ++im)
            #pragma unroll
            for (int in = 0; in < 2; ++in)
                acc[im][in] = __builtin_amdgcn_wmma_f32_16x16x32_bf16(
                    false, afr[im], false, bfr[in],
                    (short)0, acc[im][in], false, false);

        wait_async0();          // next tile fully in LDS (own wave's ops)
        __syncthreads();        // visible to all waves
    }

    // ---- epilogue: bias + activation + store fp32 ----
    #pragma unroll
    for (int im = 0; im < 4; ++im) {
        #pragma unroll
        for (int in = 0; in < 2; ++in) {
            int ncol = tileN + wn * 32 + in * 16 + lrow;
            float bv = bias[ncol];
            bool is_tanh = (ncol < HID);
            #pragma unroll
            for (int r = 0; r < 8; ++r) {
                int mrow = tileM + wm * 64 + im * 16 + half * 8 + r;
                float v = acc[im][in][r] + bv;
                v = is_tanh ? tanhf(v) : (1.0f / (1.0f + __expf(-v)));
                out[(long)mrow * NCOL + ncol] = v;
            }
        }
    }
}

// ---------------------------------------------------------------------
// fo-pooling scan. zfo holds ACTIVATED z|f|o, [S*B, 1536] f32.
// One thread per (b,h) channel. c_t = f*c + (1-f)*z ; y = o*c.
// ---------------------------------------------------------------------
__global__ void __launch_bounds__(256)
scan_kernel(const float* __restrict__ zfo,
            __bf16* __restrict__ ybf, int bf_coloff,
            float* __restrict__ yf32, int f32_coloff,
            int reverse) {
    int tid = blockIdx.x * 256 + threadIdx.x;   // 0 .. BATCH*HID-1
    int h = tid % HID;
    int b = tid / HID;
    float c = 0.0f;
    for (int t = 0; t < S_LEN; ++t) {
        int s = reverse ? (S_LEN - 1 - t) : t;
        long base = ((long)s * BATCH + b) * NCOL;
        float z = zfo[base + h];
        float f = zfo[base + HID + h];
        float o = zfo[base + 2 * HID + h];
        c = fmaf(f, c, (1.0f - f) * z);
        float y = o * c;
        if (ybf)  ybf[((long)s * BATCH + b) * Y1C + bf_coloff + h] = (__bf16)y;
        if (yf32) yf32[((long)b * S_LEN + s) * Y1C + f32_coloff + h] = y;
    }
}

// ---------------------------------------------------------------------
extern "C" void kernel_launch(void* const* d_in, const int* in_sizes, int n_in,
                              void* d_out, int out_size, void* d_ws, size_t ws_size,
                              hipStream_t stream) {
    const float* X     = (const float*)d_in[0];
    // d_in[1] = seqlens (unused, as in reference)
    const float* W_fw0 = (const float*)d_in[2];
    const float* b_fw0 = (const float*)d_in[3];
    const float* W_bw0 = (const float*)d_in[4];
    const float* b_bw0 = (const float*)d_in[5];
    const float* W_fw1 = (const float*)d_in[6];
    const float* b_fw1 = (const float*)d_in[7];
    const float* W_bw1 = (const float*)d_in[8];
    const float* b_bw1 = (const float*)d_in[9];
    float* out = (float*)d_out;

    // workspace carve-up (256B aligned)
    char* ws = (char*)d_ws;
    size_t off = 0;
    auto take = [&](size_t bytes) -> char* {
        char* p = ws + off;
        off += (bytes + 255) & ~size_t(255);
        return p;
    };
    __bf16* Xb  = (__bf16*)take((size_t)MROWS * DIN0 * 2);
    __bf16* W0f = (__bf16*)take((size_t)DIN0 * NCOL * 2);   // stored [NCOL][K]
    __bf16* W0b = (__bf16*)take((size_t)DIN0 * NCOL * 2);
    __bf16* W1f = (__bf16*)take((size_t)Y1C * NCOL * 2);
    __bf16* W1b = (__bf16*)take((size_t)Y1C * NCOL * 2);
    __bf16* Y1  = (__bf16*)take((size_t)MROWS * Y1C * 2);
    float*  ZFO = (float*)take((size_t)MROWS * NCOL * 4);
    (void)ws_size; (void)in_sizes; (void)n_in; (void)out_size;

    // 1) precision/layout conversion (weights transposed to [N][K])
    cvt_x_kernel<<<(MROWS * DIN0 / 4) / 256, 256, 0, stream>>>(X, Xb);
    dim3 gw0(DIN0 / 32, NCOL / 32), gw1(Y1C / 32, NCOL / 32);
    cvt_wt_kernel<<<gw0, 256, 0, stream>>>(W_fw0, W0f, DIN0);
    cvt_wt_kernel<<<gw0, 256, 0, stream>>>(W_bw0, W0b, DIN0);
    cvt_wt_kernel<<<gw1, 256, 0, stream>>>(W_fw1, W1f, Y1C);
    cvt_wt_kernel<<<gw1, 256, 0, stream>>>(W_bw1, W1b, Y1C);

    dim3 gg(MROWS / 128, NCOL / 128);   // 256 x 12 tiles
    int  scan_blocks = (BATCH * HID) / 256;

    // 2) layer 0 forward
    gemm_act_kernel<<<gg, 256, 0, stream>>>(Xb, W0f, b_fw0, ZFO, DIN0);
    scan_kernel<<<scan_blocks, 256, 0, stream>>>(ZFO, Y1, 0,   nullptr, 0, 0);
    // 3) layer 0 backward
    gemm_act_kernel<<<gg, 256, 0, stream>>>(Xb, W0b, b_bw0, ZFO, DIN0);
    scan_kernel<<<scan_blocks, 256, 0, stream>>>(ZFO, Y1, HID, nullptr, 0, 1);
    // 4) layer 1 forward -> final output cols [0,512)
    gemm_act_kernel<<<gg, 256, 0, stream>>>(Y1, W1f, b_fw1, ZFO, Y1C);
    scan_kernel<<<scan_blocks, 256, 0, stream>>>(ZFO, nullptr, 0, out, 0, 0);
    // 5) layer 1 backward -> final output cols [512,1024)
    gemm_act_kernel<<<gg, 256, 0, stream>>>(Y1, W1b, b_bw1, ZFO, Y1C);
    scan_kernel<<<scan_blocks, 256, 0, stream>>>(ZFO, nullptr, 0, out, HID, 1);
}
